// SelfAttention_71433896067684
// MI455X (gfx1250) — matrix operand order repaired and checked by reading, not compiled, not run on hardware
//
#include <hip/hip_runtime.h>
#include <math.h>

typedef __attribute__((ext_vector_type(16))) __bf16 v16bf;
typedef __attribute__((ext_vector_type(8)))  __bf16 v8bf;
typedef __attribute__((ext_vector_type(8)))  float  v8f;

#define SEQ 2048
#define DIM 1024

static __device__ inline v8f zero8() {
    v8f z;
    #pragma unroll
    for (int i = 0; i < 8; ++i) z[i] = 0.0f;
    return z;
}

static __device__ inline v16bf cat8(v8bf lo, v8bf hi) {
    return __builtin_shufflevector(lo, hi, 0,1,2,3,4,5,6,7,8,9,10,11,12,13,14,15);
}

static __device__ inline v8bf ld8(const __bf16* p) {
    return *(const v8bf*)p;
}

static __device__ inline v8f wmma_bf16(v16bf a, v16bf b, v8f c) {
    return __builtin_amdgcn_wmma_f32_16x16x32_bf16(
        /*neg_a=*/false, a, /*neg_b=*/false, b,
        /*c_mod=*/(short)0, c, /*reuse_a=*/false, /*reuse_b=*/false);
}

// ---------------------------------------------------------------------------
// f32 -> bf16 conversion, 8 elements / thread. n must be a multiple of 2048.
// ---------------------------------------------------------------------------
__global__ __launch_bounds__(256) void cvt_bf16_kernel(
    const float* __restrict__ src, __bf16* __restrict__ dst)
{
    const size_t i = ((size_t)blockIdx.x * 256 + threadIdx.x) * 8;
    const float4 f0 = ((const float4*)(src + i))[0];
    const float4 f1 = ((const float4*)(src + i))[1];
    v8bf r;
    r[0] = (__bf16)f0.x; r[1] = (__bf16)f0.y; r[2] = (__bf16)f0.z; r[3] = (__bf16)f0.w;
    r[4] = (__bf16)f1.x; r[5] = (__bf16)f1.y; r[6] = (__bf16)f1.z; r[7] = (__bf16)f1.w;
    *(v8bf*)(dst + i) = r;
}

// ---------------------------------------------------------------------------
// Fused QKV projection for one batch: Y = x @ W^T + b  (torch Linear).
// All inputs pre-converted to bf16.  Q,K row-major; V stored TRANSPOSED
// (Vt[e,t]) so the P@V GEMM gets k-contiguous B-fragment rows.
// Grid: (DIM/64, SEQ/64), 8 waves in 2(M) x 4(N); 32x16 tile per wave per
// output => 6 WMMAs per k-step sharing one pair of A fragments.
// ---------------------------------------------------------------------------
__global__ __launch_bounds__(256) void qkv_fused_kernel(
    const __bf16* __restrict__ x,
    const __bf16* __restrict__ Wq, const float* __restrict__ bq,
    const __bf16* __restrict__ Wk, const float* __restrict__ bk,
    const __bf16* __restrict__ Wv, const float* __restrict__ bv,
    __bf16* __restrict__ Q, __bf16* __restrict__ K, __bf16* __restrict__ Vt)
{
    const int tid  = threadIdx.x;
    const int wave = tid >> 5, lane = tid & 31;
    const int wm = wave & 1, wn = wave >> 1;
    const int row0 = blockIdx.y * 64 + wm * 32;
    const int col0 = blockIdx.x * 64 + wn * 16;
    const int lm = lane & 15, half = lane >> 4;

    const __bf16* xr0 = x  + (size_t)(row0 + lm) * DIM;
    const __bf16* xr1 = x  + (size_t)(row0 + 16 + lm) * DIM;
    const __bf16* wqr = Wq + (size_t)(col0 + lm) * DIM;
    const __bf16* wkr = Wk + (size_t)(col0 + lm) * DIM;
    const __bf16* wvr = Wv + (size_t)(col0 + lm) * DIM;

    v8f aq0 = zero8(), aq1 = zero8();
    v8f ak0 = zero8(), ak1 = zero8();
    v8f av0 = zero8(), av1 = zero8();

    for (int k0 = 0; k0 < DIM; k0 += 32) {
        // A fragments: 16x32 bf16, lane holds row M=lm, K runs half*8 and 16+half*8
        v16bf a0 = cat8(ld8(xr0 + k0 + half * 8), ld8(xr0 + k0 + 16 + half * 8));
        v16bf a1 = cat8(ld8(xr1 + k0 + half * 8), ld8(xr1 + k0 + 16 + half * 8));
        // B fragments: 32x16 bf16, lane holds col N=lm, K runs half*8 and 16+half*8
        v16bf bqf = cat8(ld8(wqr + k0 + half * 8), ld8(wqr + k0 + 16 + half * 8));
        aq0 = wmma_bf16(a0, bqf, aq0);
        aq1 = wmma_bf16(a1, bqf, aq1);
        v16bf bkf = cat8(ld8(wkr + k0 + half * 8), ld8(wkr + k0 + 16 + half * 8));
        ak0 = wmma_bf16(a0, bkf, ak0);
        ak1 = wmma_bf16(a1, bkf, ak1);
        v16bf bvf = cat8(ld8(wvr + k0 + half * 8), ld8(wvr + k0 + 16 + half * 8));
        av0 = wmma_bf16(a0, bvf, av0);
        av1 = wmma_bf16(a1, bvf, av1);
    }

    const int   c   = col0 + lm;
    const float bqv = bq[c], bkv = bk[c], bvv = bv[c];
    #pragma unroll
    for (int i = 0; i < 8; ++i) {
        const int r0 = row0 + half * 8 + i;        // C/D layout: VGPR i -> M = 8*half + i
        const int r1 = r0 + 16;
        Q[(size_t)r0 * DIM + c] = (__bf16)(aq0[i] + bqv);
        Q[(size_t)r1 * DIM + c] = (__bf16)(aq1[i] + bqv);
        K[(size_t)r0 * DIM + c] = (__bf16)(ak0[i] + bkv);
        K[(size_t)r1 * DIM + c] = (__bf16)(ak1[i] + bkv);
        Vt[(size_t)c * SEQ + r0] = (__bf16)(av0[i] + bvv);
        Vt[(size_t)c * SEQ + r1] = (__bf16)(av1[i] + bvv);
    }
}

// ---------------------------------------------------------------------------
// scores = (Q @ K^T) / sqrt(DIM), f32 out.  Block tile 128(M) x 64(N):
// 8 waves in 4(M) x 2(N), each wave owns a 32x32 tile -> 4 WMMAs per k-step
// from 2 A + 2 B fragment loads.  Grid: (SEQ/64, SEQ/128).
// ---------------------------------------------------------------------------
__global__ __launch_bounds__(256) void scores_kernel(
    const __bf16* __restrict__ Q, const __bf16* __restrict__ K,
    float* __restrict__ Sc)
{
    const int tid  = threadIdx.x;
    const int wave = tid >> 5, lane = tid & 31;
    const int wm = wave & 3, wn = wave >> 2;
    const int row0 = blockIdx.y * 128 + wm * 32;
    const int col0 = blockIdx.x * 64 + wn * 32;
    const int lm = lane & 15, half = lane >> 4;

    const __bf16* qa0 = Q + (size_t)(row0 + lm) * DIM;
    const __bf16* qa1 = Q + (size_t)(row0 + 16 + lm) * DIM;
    const __bf16* kr0 = K + (size_t)(col0 + lm) * DIM;
    const __bf16* kr1 = K + (size_t)(col0 + 16 + lm) * DIM;

    v8f a00 = zero8(), a01 = zero8(), a10 = zero8(), a11 = zero8();
    for (int k0 = 0; k0 < DIM; k0 += 32) {
        v16bf f0 = cat8(ld8(qa0 + k0 + half * 8), ld8(qa0 + k0 + 16 + half * 8));
        v16bf f1 = cat8(ld8(qa1 + k0 + half * 8), ld8(qa1 + k0 + 16 + half * 8));
        v16bf b0 = cat8(ld8(kr0 + k0 + half * 8), ld8(kr0 + k0 + 16 + half * 8));
        v16bf b1 = cat8(ld8(kr1 + k0 + half * 8), ld8(kr1 + k0 + 16 + half * 8));
        a00 = wmma_bf16(f0, b0, a00);
        a01 = wmma_bf16(f0, b1, a01);
        a10 = wmma_bf16(f1, b0, a10);
        a11 = wmma_bf16(f1, b1, a11);
    }

    const float scale = 0.03125f;  // 1/sqrt(1024)
    const int c0 = col0 + lm, c1 = c0 + 16;
    #pragma unroll
    for (int i = 0; i < 8; ++i) {
        const int r0 = row0 + half * 8 + i;
        const int r1 = r0 + 16;
        Sc[(size_t)r0 * SEQ + c0] = a00[i] * scale;
        Sc[(size_t)r0 * SEQ + c1] = a01[i] * scale;
        Sc[(size_t)r1 * SEQ + c0] = a10[i] * scale;
        Sc[(size_t)r1 * SEQ + c1] = a11[i] * scale;
    }
}

// ---------------------------------------------------------------------------
// Row-wise softmax: P[row, :] = softmax(Sc[row, :]) as bf16. One block per row.
// ---------------------------------------------------------------------------
__global__ __launch_bounds__(256) void softmax_kernel(
    const float* __restrict__ Sc, __bf16* __restrict__ P)
{
    __shared__ float red[256];
    const int row = blockIdx.x, tid = threadIdx.x;
    const float* sr = Sc + (size_t)row * SEQ;

    float m = -3.0e38f;
    for (int j = tid; j < SEQ; j += 256) m = fmaxf(m, sr[j]);
    red[tid] = m; __syncthreads();
    for (int s = 128; s > 0; s >>= 1) {
        if (tid < s) red[tid] = fmaxf(red[tid], red[tid + s]);
        __syncthreads();
    }
    m = red[0]; __syncthreads();

    float sum = 0.0f;
    for (int j = tid; j < SEQ; j += 256) sum += __expf(sr[j] - m);
    red[tid] = sum; __syncthreads();
    for (int s = 128; s > 0; s >>= 1) {
        if (tid < s) red[tid] += red[tid + s];
        __syncthreads();
    }
    const float inv = 1.0f / red[0];

    __bf16* pr = P + (size_t)row * SEQ;
    for (int j = tid; j < SEQ; j += 256)
        pr[j] = (__bf16)(__expf(sr[j] - m) * inv);
}

// ---------------------------------------------------------------------------
// out (+)= (P @ V) / 8 using Vt.  Same 128x64 / 32x32-per-wave tiling.
// Grid: (DIM/64, SEQ/128).  beta==0 (batch 0) overwrites (poison/replay safe).
// ---------------------------------------------------------------------------
__global__ __launch_bounds__(256) void av_kernel(
    const __bf16* __restrict__ P, const __bf16* __restrict__ Vt,
    float* __restrict__ out, int beta)
{
    const int tid  = threadIdx.x;
    const int wave = tid >> 5, lane = tid & 31;
    const int wm = wave & 3, wn = wave >> 2;
    const int row0 = blockIdx.y * 128 + wm * 32;
    const int col0 = blockIdx.x * 64 + wn * 32;
    const int lm = lane & 15, half = lane >> 4;

    const __bf16* pa0 = P + (size_t)(row0 + lm) * SEQ;
    const __bf16* pa1 = P + (size_t)(row0 + 16 + lm) * SEQ;
    const __bf16* vr0 = Vt + (size_t)(col0 + lm) * SEQ;
    const __bf16* vr1 = Vt + (size_t)(col0 + 16 + lm) * SEQ;

    v8f a00 = zero8(), a01 = zero8(), a10 = zero8(), a11 = zero8();
    for (int k0 = 0; k0 < SEQ; k0 += 32) {
        v16bf f0 = cat8(ld8(pa0 + k0 + half * 8), ld8(pa0 + k0 + 16 + half * 8));
        v16bf f1 = cat8(ld8(pa1 + k0 + half * 8), ld8(pa1 + k0 + 16 + half * 8));
        v16bf b0 = cat8(ld8(vr0 + k0 + half * 8), ld8(vr0 + k0 + 16 + half * 8));
        v16bf b1 = cat8(ld8(vr1 + k0 + half * 8), ld8(vr1 + k0 + 16 + half * 8));
        a00 = wmma_bf16(f0, b0, a00);
        a01 = wmma_bf16(f0, b1, a01);
        a10 = wmma_bf16(f1, b0, a10);
        a11 = wmma_bf16(f1, b1, a11);
    }

    const int c0 = col0 + lm, c1 = c0 + 16;
    #pragma unroll
    for (int i = 0; i < 8; ++i) {
        const int r0 = row0 + half * 8 + i;
        const int r1 = r0 + 16;
        const size_t i00 = (size_t)r0 * DIM + c0;
        const size_t i01 = (size_t)r0 * DIM + c1;
        const size_t i10 = (size_t)r1 * DIM + c0;
        const size_t i11 = (size_t)r1 * DIM + c1;
        const float v00 = a00[i] * 0.125f;  // mean over 8 batches
        const float v01 = a01[i] * 0.125f;
        const float v10 = a10[i] * 0.125f;
        const float v11 = a11[i] * 0.125f;
        out[i00] = beta ? (out[i00] + v00) : v00;
        out[i01] = beta ? (out[i01] + v01) : v01;
        out[i10] = beta ? (out[i10] + v10) : v10;
        out[i11] = beta ? (out[i11] + v11) : v11;
    }
}

// ---------------------------------------------------------------------------
// Host: weights converted to bf16 once; batches processed sequentially on
// `stream` reusing 46 MB of d_ws:
//   [0,2M)    Wq bf16      [2M,4M)   Wk bf16     [4M,6M)  Wv bf16
//   [6M,10M)  xb bf16 [SEQ,DIM]
//   [10M,14M) Q   [14M,18M) K   [18M,22M) Vt (transposed, [DIM,SEQ])
//   [22M,38M) scores f32 [SEQ,SEQ]
//   [38M,46M) P bf16 [SEQ,SEQ]
// ---------------------------------------------------------------------------
extern "C" void kernel_launch(void* const* d_in, const int* in_sizes, int n_in,
                              void* d_out, int out_size, void* d_ws, size_t ws_size,
                              hipStream_t stream) {
    const float* x  = (const float*)d_in[0];
    const float* Wq = (const float*)d_in[1];
    const float* bq = (const float*)d_in[2];
    const float* Wk = (const float*)d_in[3];
    const float* bk = (const float*)d_in[4];
    const float* Wv = (const float*)d_in[5];
    const float* bv = (const float*)d_in[6];
    float* out = (float*)d_out;

    char* ws = (char*)d_ws;
    __bf16* Wqb = (__bf16*)(ws);
    __bf16* Wkb = (__bf16*)(ws + (2u  << 20));
    __bf16* Wvb = (__bf16*)(ws + (4u  << 20));
    __bf16* xb  = (__bf16*)(ws + (6u  << 20));
    __bf16* Q   = (__bf16*)(ws + (10u << 20));
    __bf16* K   = (__bf16*)(ws + (14u << 20));
    __bf16* Vt  = (__bf16*)(ws + (18u << 20));
    float*  Sc  = (float*) (ws + (22u << 20));
    __bf16* P   = (__bf16*)(ws + (38u << 20));

    const dim3 blk(256);
    const int wElems = DIM * DIM;            // 1M
    const int xElems = SEQ * DIM;            // 2M
    cvt_bf16_kernel<<<dim3(wElems / 2048), blk, 0, stream>>>(Wq, Wqb);
    cvt_bf16_kernel<<<dim3(wElems / 2048), blk, 0, stream>>>(Wk, Wkb);
    cvt_bf16_kernel<<<dim3(wElems / 2048), blk, 0, stream>>>(Wv, Wvb);

    for (int b = 0; b < 8; ++b) {
        cvt_bf16_kernel<<<dim3(xElems / 2048), blk, 0, stream>>>(
            x + (size_t)b * xElems, xb);
        qkv_fused_kernel<<<dim3(DIM / 64, SEQ / 64), blk, 0, stream>>>(
            xb, Wqb, bq, Wkb, bk, Wvb, bv, Q, K, Vt);
        scores_kernel<<<dim3(SEQ / 64, SEQ / 128), blk, 0, stream>>>(Q, K, Sc);
        softmax_kernel<<<dim3(SEQ), blk, 0, stream>>>(Sc, P);
        av_kernel<<<dim3(DIM / 64, SEQ / 128), blk, 0, stream>>>(P, Vt, out, b > 0 ? 1 : 0);
    }
}
